// QLSTM_65481071397059
// MI455X (gfx1250) — matrix operand order repaired and checked by reading, not compiled, not run on hardware
//
#include <hip/hip_runtime.h>

typedef __attribute__((ext_vector_type(16))) _Float16 v16h;
typedef __attribute__((ext_vector_type(8)))  float    v8f;

#define SEQ     512
#define BATCH   64
#define IN_DIM  512
#define HID     512
#define NGATE   2048           // 4 * HID
#define NUM_WG  8
#define THREADS 256

// ---- workspace layout (bytes) ----
#define CNT_OFF    0
#define XPACK_OFF  256
#define XPACK_HALF (SEQ * 4 * 16 * 512)        // 16,777,216 halfs (32 MB): [s][mt4][kt16][512]
#define WPACK_OFF  (XPACK_OFF + XPACK_HALF * 2)
#define WPACK_HALF (128 * 32 * 512)            // 2,097,152 halfs (4 MB): [nt128][kt32][512]
#define BIAS_OFF   (WPACK_OFF + WPACK_HALF * 2)
#define HPACK_OFF  (BIAS_OFF + NGATE * 4)
#define HPACK_HALF (4 * 16 * 512)              // 32768 halfs per buffer: [mt4][kt16][512]

#define GB_STRIDE 68                            // padded LDS row stride (bank-conflict mitigation)
#define GB_GATE   (64 * GB_STRIDE)              // floats per gate
#define LDS_BYTES (4 * GB_GATE * 4)             // 69632 bytes

__device__ __forceinline__ float sigf(float x) { return 1.0f / (1.0f + __expf(-x)); }

// ---------- pack x into WMMA A-fragment layout (f16) ----------
// A 16x32 tile, 512 halfs: idx = lane*16 + slot
//   m = lane%16; hi = lane/16; slot<8 -> k = hi*8+slot ; slot>=8 -> k = 16+hi*8+(slot-8)
__global__ void k_pack_x(const float* __restrict__ x, char* __restrict__ ws) {
    const size_t idx = (size_t)blockIdx.x * blockDim.x + threadIdx.x;
    if (idx >= (size_t)XPACK_HALF) return;
    _Float16* xp = (_Float16*)(ws + XPACK_OFF);
    const int r    = (int)(idx & 511);
    const int tile = (int)(idx >> 9);
    const int lane = r >> 4;
    const int slot = r & 15;
    const int kt   = tile & 15;
    const int mt   = (tile >> 4) & 3;
    const int s    = tile >> 6;
    const int m16  = lane & 15, hi = lane >> 4;
    const int k    = (slot < 8) ? (hi * 8 + slot) : (16 + hi * 8 + (slot - 8));
    const int b    = mt * 16 + m16;
    const int kg   = kt * 32 + k;
    xp[idx] = (_Float16)x[((size_t)s * BATCH + b) * IN_DIM + kg];
}

// ---------- pack weights (rows 0..511 = x-part, 512..1023 = h-part) + bias ----------
// B 32x16 tile, 512 halfs: idx = lane*16 + slot ; n = lane%16 ; slot<8 -> k = (lane/16)*8+slot ; slot>=8 -> k = 16+(lane/16)*8+(slot-8)
__global__ void k_pack_w(const float* __restrict__ Wf, const float* __restrict__ Wi,
                         const float* __restrict__ Wu, const float* __restrict__ Wo,
                         const float* __restrict__ bf, const float* __restrict__ bi,
                         const float* __restrict__ bu, const float* __restrict__ bo,
                         char* __restrict__ ws) {
    const size_t idx = (size_t)blockIdx.x * blockDim.x + threadIdx.x;
    if (idx < (size_t)WPACK_HALF) {
        _Float16* wp = (_Float16*)(ws + WPACK_OFF);
        const int r    = (int)(idx & 511);
        const int tile = (int)(idx >> 9);
        const int lane = r >> 4;
        const int slot = r & 15;
        const int nt   = tile >> 5;
        const int kt   = tile & 31;
        const int n    = nt * 16 + (lane & 15);
        const int kg   = kt * 32 + ((slot < 8) ? ((lane >> 4) * 8 + slot) : (16 + (lane >> 4) * 8 + (slot - 8)));   // 0..1023
        const int g    = n >> 9, j = n & 511;
        const float* W = (g == 0) ? Wf : (g == 1) ? Wi : (g == 2) ? Wu : Wo;
        wp[idx] = (_Float16)W[(size_t)kg * HID + j];
    } else if (idx < (size_t)WPACK_HALF + NGATE) {
        const int e = (int)(idx - WPACK_HALF);
        const int g = e >> 9, j = e & 511;
        const float* B = (g == 0) ? bf : (g == 1) ? bi : (g == 2) ? bu : bo;
        ((float*)(ws + BIAS_OFF))[e] = B[j];
    }
}

// ---------- per-launch init: sync counter + h0 pack = 0 ----------
__global__ void k_init(char* __restrict__ ws) {
    const int idx = blockIdx.x * blockDim.x + threadIdx.x;
    if (idx == 0) *(int*)(ws + CNT_OFF) = 0;
    if (idx < 2 * HPACK_HALF / 2)   // both h buffers, as dwords
        ((unsigned int*)(ws + HPACK_OFF))[idx] = 0u;
}

// ---------- persistent recurrent LSTM kernel: 8 WGs x 8 waves, 512 steps ----------
__global__ __launch_bounds__(THREADS, 1)
void k_lstm(char* __restrict__ ws, float* __restrict__ out) {
    extern __shared__ __align__(16) float gbuf[];            // [4 gates][64 cols][68]
    const _Float16* xp    = (const _Float16*)(ws + XPACK_OFF);
    const _Float16* wp    = (const _Float16*)(ws + WPACK_OFF);
    const float*    biasb = (const float*)(ws + BIAS_OFF);
    _Float16*       hp    = (_Float16*)(ws + HPACK_OFF);
    int*            cnt   = (int*)(ws + CNT_OFF);

    const int tid  = threadIdx.x;
    const int wg   = blockIdx.x;          // owns hid cols [wg*64, wg*64+64)
    const int wave = tid >> 5;
    const int lane = tid & 31;
    const int l15  = lane & 15;
    const int hi8  = (lane >> 4) << 3;

    // wave -> fixed batch tile + gate pair; A fragment shared across its 8 N-tiles
    const int mt    = wave >> 1;          // batch tile 0..3
    const int gbase = (wave & 1) * 2;     // gates {0,1} or {2,3}

    float creg[16];                        // cell state, element e = tid + 256*i
#pragma unroll
    for (int i = 0; i < 16; ++i) creg[i] = 0.0f;

#pragma unroll 1
    for (int s = 0; s < SEQ; ++s) {
        const _Float16* hpR = hp + (size_t)(s & 1) * HPACK_HALF;
        _Float16*       hpW = hp + (size_t)((s + 1) & 1) * HPACK_HALF;

        // ---- gate GEMM: load A once per kt, fan out to 8 independent accumulators ----
        {
            const _Float16* ax = xp  + ((size_t)(s * 4 + mt) * 16) * 512 + lane * 16;
            const _Float16* ah = hpR + (size_t)mt * 16 * 512 + lane * 16;
            // one base for all 8 B tiles; per-tile delta fits the 24-bit imm offset
            const _Float16* bb = wp + (size_t)(gbase * 32 + wg * 4) * (32 * 512) + lane * 16;
            v8f acc[8];
#pragma unroll
            for (int u = 0; u < 8; ++u) acc[u] = (v8f){};

#pragma unroll 2
            for (int kt = 0; kt < 16; ++kt) {     // x contribution (K = 0..511)
                v16h a = *(const v16h*)(ax + (size_t)kt * 512);
                const _Float16* bk = bb + (size_t)kt * 512;
#pragma unroll
                for (int u = 0; u < 8; ++u) {
                    v16h b = *(const v16h*)(bk + (size_t)((u & 3) + (u >> 2) * 32) * (32 * 512));
                    acc[u] = __builtin_amdgcn_wmma_f32_16x16x32_f16(false, a, false, b,
                                                                    (short)0, acc[u],
                                                                    false, false);
                }
            }
#pragma unroll 2
            for (int kt = 0; kt < 16; ++kt) {     // h contribution (K = 512..1023)
                v16h a = *(const v16h*)(ah + (size_t)kt * 512);
                const _Float16* bk = bb + (size_t)(16 + kt) * 512;
#pragma unroll
                for (int u = 0; u < 8; ++u) {
                    v16h b = *(const v16h*)(bk + (size_t)((u & 3) + (u >> 2) * 32) * (32 * 512));
                    acc[u] = __builtin_amdgcn_wmma_f32_16x16x32_f16(false, a, false, b,
                                                                    (short)0, acc[u],
                                                                    false, false);
                }
            }
            // epilogue: bias + scatter to LDS gate buffer
#pragma unroll
            for (int u = 0; u < 8; ++u) {
                const int g   = gbase + (u >> 2);
                const int jt  = u & 3;
                const int ntg = g * 32 + wg * 4 + jt;
                const float bv = biasb[ntg * 16 + l15];
                float* go = gbuf + g * GB_GATE + (jt * 16 + l15) * GB_STRIDE;
#pragma unroll
                for (int r = 0; r < 8; ++r)       // C layout: VGPR r -> M = r (+8 hi half)
                    go[mt * 16 + hi8 + r] = acc[u][r] + bv;
            }
        }
        __syncthreads();

        // ---- elementwise: c/h update, write output h, repack h to f16 fragments ----
#pragma unroll 1
        for (int i = 0; i < 16; ++i) {
            const int e  = tid + (i << 8);        // 0..4095
            const int cl = e >> 6;                // local col 0..63
            const int m  = e & 63;                // batch 0..63
            const int o0 = cl * GB_STRIDE + m;
            const float fg = sigf(gbuf[o0]);
            const float ig = sigf(gbuf[GB_GATE + o0]);
            const float gg = tanhf(gbuf[2 * GB_GATE + o0]);
            const float og = sigf(gbuf[3 * GB_GATE + o0]);
            const float cn = fg * creg[i] + ig * gg;
            creg[i] = cn;
            const float h = og * tanhf(cn);
            const int jg = wg * 64 + cl;
            out[((size_t)s * BATCH + m) * HID + jg] = h;
            // scatter into next-step A-fragment layout
            const int mt2 = m >> 4, m16 = m & 15;
            const int kt2 = jg >> 5, k32 = jg & 31;
            int hii, slot;
            if (k32 < 16) { hii = k32 >> 3; slot = k32 & 7; }
            else          { const int kk = k32 - 16; hii = kk >> 3; slot = 8 + (kk & 7); }
            hpW[(size_t)(mt2 * 16 + kt2) * 512 + (hii * 16 + m16) * 16 + slot] = (_Float16)h;
            if (s == SEQ - 1) {
                out[(size_t)SEQ * BATCH * HID + (size_t)m * HID + jg] = h;   // hx
                out[(size_t)SEQ * BATCH * HID + (size_t)BATCH * HID
                    + (size_t)m * HID + jg] = cn;                            // cx
            }
        }
        __threadfence();
        __syncthreads();

        // ---- grid-wide barrier (8 co-resident WGs) ----
        if (tid == 0) {
            __hip_atomic_fetch_add(cnt, 1, __ATOMIC_RELEASE, __HIP_MEMORY_SCOPE_AGENT);
            const int target = NUM_WG * (s + 1);
            while (__hip_atomic_load(cnt, __ATOMIC_ACQUIRE, __HIP_MEMORY_SCOPE_AGENT) < target)
                __builtin_amdgcn_s_sleep(1);
        }
        __syncthreads();
        __builtin_amdgcn_fence(__ATOMIC_ACQUIRE, "agent");
    }
}

extern "C" void kernel_launch(void* const* d_in, const int* in_sizes, int n_in,
                              void* d_out, int out_size, void* d_ws, size_t ws_size,
                              hipStream_t stream) {
    (void)in_sizes; (void)n_in; (void)out_size; (void)ws_size;
    const float* inp = (const float*)d_in[0];
    const float* Wf  = (const float*)d_in[1];
    const float* bf  = (const float*)d_in[2];
    const float* Wi  = (const float*)d_in[3];
    const float* bi  = (const float*)d_in[4];
    const float* Wu  = (const float*)d_in[5];
    const float* bu  = (const float*)d_in[6];
    const float* Wo  = (const float*)d_in[7];
    const float* bo  = (const float*)d_in[8];
    char* ws = (char*)d_ws;
    float* out = (float*)d_out;

    (void)hipFuncSetAttribute((const void*)k_lstm,
                              hipFuncAttributeMaxDynamicSharedMemorySize, LDS_BYTES);

    k_init<<<(2 * HPACK_HALF / 2 + THREADS - 1) / THREADS, THREADS, 0, stream>>>(ws);
    k_pack_w<<<(WPACK_HALF + NGATE + THREADS - 1) / THREADS, THREADS, 0, stream>>>(
        Wf, Wi, Wu, Wo, bf, bi, bu, bo, ws);
    k_pack_x<<<XPACK_HALF / THREADS, THREADS, 0, stream>>>(inp, ws);
    k_lstm<<<NUM_WG, THREADS, LDS_BYTES, stream>>>(ws, out);
}
